// MSGT_7438883357616
// MI455X (gfx1250) — matrix operand order repaired and checked by reading, not compiled, not run on hardware
//
#include <hip/hip_runtime.h>
#include <math.h>

// ---------------------------------------------------------------------------
// Transformer encoder fwd (B=64,S=512,D=256,L=3) for MI455X (gfx1250).
// ~129 GFLOP vs ~0.2 GB traffic -> compute-bound on matrix units:
// v_wmma_f32_16x16x32_f16 everywhere, f32 accumulate.
// Weights (and the V activation) are stored pre-swizzled in B-fragment order
// so every WMMA B operand is a single contiguous 32B read per lane.
// ---------------------------------------------------------------------------

typedef __attribute__((ext_vector_type(16))) _Float16 v16h;
typedef __attribute__((ext_vector_type(8)))  _Float16 v8h;
typedef __attribute__((ext_vector_type(8)))  float    v8f;

#define BB 64
#define SS 512
#define DD 256
#define LL 3
#define NTOK (BB * SS)

__device__ inline v8f wmma_f16(v16h a, v16h b, v8f c) {
  return __builtin_amdgcn_wmma_f32_16x16x32_f16(
      /*neg_a=*/false, a, /*neg_b=*/false, b,
      /*c_mod=*/(short)0, c, /*reuse_a=*/false, /*reuse_b=*/false);
}

// ---- fragment loaders (this file's contiguous k order k = 16*(lane/16) + e, same for A and B)
// A matrix 16x32 (MxK), row-major source. Per lane: two contiguous 16B chunks
// (K = kb..kb+7 and kb+8..kb+15, kb = 16 for lanes 16-31) -> 2x b128.
__device__ inline v16h load_a_rm(const _Float16* A, int lda, int lane) {
  const _Float16* p =
      A + (size_t)(lane & 15) * lda + ((lane & 16) ? 16 : 0);
  v8h lo = *(const v8h*)(p);
  v8h hi = *(const v8h*)(p + 8);
  v16h r;
#pragma unroll
  for (int e = 0; e < 8; ++e) { r[e] = lo[e]; r[e + 8] = hi[e]; }
  return r;
}

// B matrix 32x16 tile from pre-swizzled storage: lane's 16 halves contiguous.
// Tile t occupies 512 halves at Bsw + t*512; lane L's data at +L*16.
__device__ inline v16h load_b_swz(const v16h* Bsw, size_t tile, int lane) {
  return Bsw[tile * 32 + lane];
}

// B matrix 32x16 = K^T chunk from row-major K[S x D]: contiguous per lane.
__device__ inline v16h load_bT(const _Float16* Kd, int ldk, int lane) {
  return *(const v16h*)(Kd + (size_t)(lane & 15) * ldk +
                        ((lane & 16) ? 16 : 0));
}

__device__ inline float gelu_exact(float x) {
  return 0.5f * x * (1.0f + erff(x * 0.70710678118654752f));
}

// ---------------------------------------------------------------------------
// Setup kernels
// ---------------------------------------------------------------------------
// Convert f32 weights [L, K, N] into f16 B-fragment-swizzled order:
// out[((j*K/32 + kt)*(N/16) + nt)*512 + l*16 + e] = W[j, kt*32 + (l&16?16:0)+e,
//                                                      nt*16 + (l&15)]
__global__ void pack_bswz_kernel(const float* __restrict__ W,
                                 _Float16* __restrict__ out,
                                 int Kdim, int Ndim, int total) {
  int i = blockIdx.x * blockDim.x + threadIdx.x;
  if (i >= total) return;
  const int per_layer = Kdim * Ndim;
  const int j = i / per_layer;
  const int il = i - j * per_layer;
  const int e = il & 15;
  const int l = (il >> 4) & 31;
  const int tile = il >> 9;
  const int nt = tile % (Ndim >> 4);
  const int kt = tile / (Ndim >> 4);
  const int Ki = kt * 32 + (l & 16) + e;    // (l&16) is 0 or 16
  const int Ni = nt * 16 + (l & 15);
  out[i] = (_Float16)W[(size_t)j * per_layer + (size_t)Ki * Ndim + Ni];
}

__global__ void relpos_bias_kernel(const float* __restrict__ table,
                                   float* __restrict__ cp, int S) {
  int i = blockIdx.x * blockDim.x + threadIdx.x;
  if (i >= S * S) return;
  int q = i / S, k = i % S;
  int rel = k - q;                          // memory - context
  int bucket = (rel < 0) ? (-rel + S - 1) : rel;
  cp[i] = table[bucket];
}

// ---------------------------------------------------------------------------
// LayerNorm: one 256-thread block per D=256 row. Emits f32 + f16 copies.
// ---------------------------------------------------------------------------
__global__ __launch_bounds__(DD) void ln_kernel(
    const float* __restrict__ x, const float* __restrict__ g,
    const float* __restrict__ b, float* __restrict__ h,
    _Float16* __restrict__ h16) {
  const int row = blockIdx.x;
  const int t = threadIdx.x;
  const float v = x[(size_t)row * DD + t];
  __shared__ float s1[8], s2[8];
  float sum = v, sq = v * v;
#pragma unroll
  for (int o = 16; o > 0; o >>= 1) {
    sum += __shfl_xor(sum, o, 32);
    sq  += __shfl_xor(sq,  o, 32);
  }
  if ((t & 31) == 0) { s1[t >> 5] = sum; s2[t >> 5] = sq; }
  __syncthreads();
  float tot = 0.f, totq = 0.f;
#pragma unroll
  for (int i = 0; i < 8; ++i) { tot += s1[i]; totq += s2[i]; }
  const float mean = tot * (1.0f / DD);
  const float var  = totq * (1.0f / DD) - mean * mean;
  const float hv = (v - mean) * rsqrtf(var + 1e-5f) * g[t] + b[t];
  h[(size_t)row * DD + t] = hv;
  h16[(size_t)row * DD + t] = (_Float16)hv;
}

// xnew = h + LN_j(oproj); c16 = LN_cm(xnew)  (two chained row-norms fused)
__global__ __launch_bounds__(DD) void resid_ln_kernel(
    const float* __restrict__ h, const float* __restrict__ op,
    const float* __restrict__ g, const float* __restrict__ b,
    const float* __restrict__ cg, const float* __restrict__ cb,
    float* __restrict__ xnew, _Float16* __restrict__ c16) {
  const int row = blockIdx.x;
  const int t = threadIdx.x;
  __shared__ float s1[8], s2[8];

  float v = op[(size_t)row * DD + t];
  float sum = v, sq = v * v;
#pragma unroll
  for (int o = 16; o > 0; o >>= 1) {
    sum += __shfl_xor(sum, o, 32);
    sq  += __shfl_xor(sq,  o, 32);
  }
  if ((t & 31) == 0) { s1[t >> 5] = sum; s2[t >> 5] = sq; }
  __syncthreads();
  float tot = 0.f, totq = 0.f;
#pragma unroll
  for (int i = 0; i < 8; ++i) { tot += s1[i]; totq += s2[i]; }
  float mean = tot * (1.0f / DD);
  float var  = totq * (1.0f / DD) - mean * mean;
  float xn = h[(size_t)row * DD + t] +
             (v - mean) * rsqrtf(var + 1e-5f) * g[t] + b[t];
  xnew[(size_t)row * DD + t] = xn;
  __syncthreads();

  sum = xn; sq = xn * xn;
#pragma unroll
  for (int o = 16; o > 0; o >>= 1) {
    sum += __shfl_xor(sum, o, 32);
    sq  += __shfl_xor(sq,  o, 32);
  }
  if ((t & 31) == 0) { s1[t >> 5] = sum; s2[t >> 5] = sq; }
  __syncthreads();
  tot = 0.f; totq = 0.f;
#pragma unroll
  for (int i = 0; i < 8; ++i) { tot += s1[i]; totq += s2[i]; }
  mean = tot * (1.0f / DD);
  var  = totq * (1.0f / DD) - mean * mean;
  c16[(size_t)row * DD + t] =
      (_Float16)((xn - mean) * rsqrtf(var + 1e-5f) * cg[t] + cb[t]);
}

// ---------------------------------------------------------------------------
// WMMA GEMM: C[M,N] = A[M,K] * B[K,N] + bias (+gelu) (+residual).
// B is pre-swizzled (fragment-contiguous). One wave owns a 16x64 strip
// (A-frag reused across 4 independent accumulators). 8 waves/block.
// OUT_MODE: 0 = f32 row-major, 1 = f16 row-major,
//           2 = f16 swizzled-B layout per SS-row batch (for V).
// ---------------------------------------------------------------------------
template <int OUT_MODE, int ACT_GELU, int RES>
__global__ __launch_bounds__(256) void gemm_wmma_kernel(
    const _Float16* __restrict__ A, const _Float16* __restrict__ Bw,
    const float* __restrict__ bias, const float* __restrict__ res,
    void* __restrict__ out, int M, int N, int K) {
  const int lane = threadIdx.x & 31;
  const int wid = blockIdx.x * (blockDim.x >> 5) + (threadIdx.x >> 5);
  const int ntn = N >> 6;                    // 64-wide N strips
  const int mt = wid / ntn;
  const int nt = wid % ntn;
  if (mt * 16 >= M) return;                  // wave-uniform, EXEC stays full

  v8f acc0 = {}, acc1 = {}, acc2 = {}, acc3 = {};
  const _Float16* Arow = A + (size_t)mt * 16 * K;
  const v16h* Bsw = (const v16h*)Bw;
  const int ntiles16 = N >> 4;
#pragma unroll 2
  for (int k0 = 0; k0 < K; k0 += 32) {
    const size_t tb = (size_t)(k0 >> 5) * ntiles16 + nt * 4;
    v16h a  = load_a_rm(Arow + k0, K, lane);
    v16h b0 = load_b_swz(Bsw, tb + 0, lane);
    v16h b1 = load_b_swz(Bsw, tb + 1, lane);
    v16h b2 = load_b_swz(Bsw, tb + 2, lane);
    v16h b3 = load_b_swz(Bsw, tb + 3, lane);
    acc0 = wmma_f16(a, b0, acc0);
    acc1 = wmma_f16(a, b1, acc1);
    acc2 = wmma_f16(a, b2, acc2);
    acc3 = wmma_f16(a, b3, acc3);
  }

  const int mbase = mt * 16 + ((lane & 16) ? 8 : 0);
  const int ncol = lane & 15;
  v8f accs[4] = {acc0, acc1, acc2, acc3};
#pragma unroll
  for (int i = 0; i < 4; ++i) {
    const int n = nt * 64 + i * 16 + ncol;
    const float bv = bias ? bias[n] : 0.0f;
#pragma unroll
    for (int r = 0; r < 8; ++r) {
      const int m = mbase + r;
      float v = accs[i][r] + bv;
      if (ACT_GELU) v = gelu_exact(v);
      if (RES) v += res[(size_t)m * N + n];
      if (OUT_MODE == 0) {
        ((float*)out)[(size_t)m * N + n] = v;
      } else if (OUT_MODE == 1) {
        ((_Float16*)out)[(size_t)m * N + n] = (_Float16)v;
      } else {
        // write into per-batch swizzled B layout: batch = m/SS, k-row = m%SS
        const int bidx = m >> 9;             // / SS
        const int s = m & (SS - 1);
        const int kr = s & 31;
        const int l2 = ncol + (kr & 16);
        const int e2 = kr & 15;
        const size_t off =
            (size_t)bidx * SS * DD +
            ((size_t)((s >> 5) * (DD >> 4) + (n >> 4)) * 32 + l2) * 16 + e2;
        ((_Float16*)out)[off] = (_Float16)v;
      }
    }
  }
}

// ---------------------------------------------------------------------------
// Attention: one wave per (batch, 16-query tile). Scores in LDS, wave32
// shuffle softmax, P@V with WMMA (A-frags from LDS, V pre-swizzled).
// ---------------------------------------------------------------------------
__global__ __launch_bounds__(32) void attn_kernel(
    const _Float16* __restrict__ Q, const _Float16* __restrict__ Kmat,
    const _Float16* __restrict__ Vsw, const float* __restrict__ cp,
    const int* __restrict__ mask, _Float16* __restrict__ O,
    float inv_scale) {
  __shared__ float    sc[16 * SS];          // 32 KB scores
  __shared__ _Float16 pp[16 * SS];          // 16 KB probabilities
  const int qt = blockIdx.x;
  const int b = blockIdx.y;
  const int lane = threadIdx.x;
  const _Float16* Qb = Q + ((size_t)b * SS + qt * 16) * DD;
  const _Float16* Kb = Kmat + (size_t)b * SS * DD;
  const v16h* Vb = (const v16h*)Vsw + (size_t)b * (SS * DD / 16);
  const int mbase = (lane & 16) ? 8 : 0;
  const int ncol = lane & 15;

  // Q A-fragments are loop-invariant: preload all 8 (D=256 / K=32).
  v16h aq[8];
#pragma unroll
  for (int kc = 0; kc < 8; ++kc) aq[kc] = load_a_rm(Qb + kc * 32, DD, lane);

  // --- phase 1: S = Q K^T / sqrt(D) + cp, with key padding mask -----------
  for (int kt = 0; kt < SS / 16; ++kt) {
    // prefetch next K tile into near cache (locality 3 -> WGP scope:
    // fills every cache level on a miss); 8KB tile, 2 lines per lane
    if (kt + 1 < SS / 16) {
      const char* nk = (const char*)(Kb + (size_t)(kt + 1) * 16 * DD);
      __builtin_prefetch(nk + lane * 256, 0, 3);
      __builtin_prefetch(nk + lane * 256 + 128, 0, 3);
    }
    const _Float16* krow = Kb + (size_t)(kt * 16) * DD;
    v16h bf[8];
#pragma unroll
    for (int kc = 0; kc < 8; ++kc) bf[kc] = load_bT(krow + kc * 32, DD, lane);
    v8f a0 = {}, a1 = {};
#pragma unroll
    for (int kc = 0; kc < 8; kc += 2) {
      a0 = wmma_f16(aq[kc], bf[kc], a0);
      a1 = wmma_f16(aq[kc + 1], bf[kc + 1], a1);
    }
    v8f acc = a0 + a1;
    const int kk = kt * 16 + ncol;
    const bool masked = (mask[b * SS + kk] == 0);
    const float* cprow = cp + (size_t)(qt * 16) * SS + kk;
#pragma unroll
    for (int r = 0; r < 8; ++r) {
      const int m = mbase + r;
      const float cpv = cprow[(size_t)m * SS];
      const float vv = fmaf(acc[r], inv_scale, cpv);
      sc[m * SS + kk] = masked ? -3.4028235e38f : vv;
    }
  }
  __syncthreads();

  // --- phase 2: row softmax ------------------------------------------------
  for (int m = 0; m < 16; ++m) {
    float mx = -3.4028235e38f;
    for (int i = lane; i < SS; i += 32) mx = fmaxf(mx, sc[m * SS + i]);
#pragma unroll
    for (int o = 16; o > 0; o >>= 1) mx = fmaxf(mx, __shfl_xor(mx, o, 32));
    float sum = 0.f;
    for (int i = lane; i < SS; i += 32) {
      float e = __expf(sc[m * SS + i] - mx);
      sc[m * SS + i] = e;
      sum += e;
    }
#pragma unroll
    for (int o = 16; o > 0; o >>= 1) sum += __shfl_xor(sum, o, 32);
    const float inv = 1.0f / sum;
    for (int i = lane; i < SS; i += 32)
      pp[m * SS + i] = (_Float16)(sc[m * SS + i] * inv);
  }
  __syncthreads();

  // --- phase 3: O = P V (V pre-swizzled, 2 independent accumulators) ------
  for (int nt = 0; nt < DD / 16; ++nt) {
    v8f a0 = {}, a1 = {};
#pragma unroll 2
    for (int k0 = 0; k0 < SS; k0 += 64) {
      v16h pa0 = load_a_rm(pp + k0, SS, lane);
      v16h pa1 = load_a_rm(pp + k0 + 32, SS, lane);
      v16h b0 = Vb[((size_t)(k0 >> 5) * (DD >> 4) + nt) * 32 + lane];
      v16h b1 = Vb[((size_t)((k0 >> 5) + 1) * (DD >> 4) + nt) * 32 + lane];
      a0 = wmma_f16(pa0, b0, a0);
      a1 = wmma_f16(pa1, b1, a1);
    }
    v8f acc = a0 + a1;
    const int n = nt * 16 + ncol;
#pragma unroll
    for (int r = 0; r < 8; ++r)
      O[((size_t)b * SS + qt * 16 + mbase + r) * DD + n] = (_Float16)acc[r];
  }
}

// ---------------------------------------------------------------------------
// Host-side orchestration
// ---------------------------------------------------------------------------
extern "C" void kernel_launch(void* const* d_in, const int* in_sizes, int n_in,
                              void* d_out, int out_size, void* d_ws,
                              size_t ws_size, hipStream_t stream) {
  (void)in_sizes; (void)n_in; (void)out_size; (void)ws_size;
  const float* x_in  = (const float*)d_in[0];
  const int*   mask  = (const int*)d_in[1];
  const float* table = (const float*)d_in[2];
  const float* ln_g  = (const float*)d_in[3];
  const float* ln_b  = (const float*)d_in[4];
  const float* wq    = (const float*)d_in[5];
  const float* bq    = (const float*)d_in[6];
  const float* wk    = (const float*)d_in[7];
  const float* bk    = (const float*)d_in[8];
  const float* wv    = (const float*)d_in[9];
  const float* bv    = (const float*)d_in[10];
  const float* wo    = (const float*)d_in[11];
  const float* bo    = (const float*)d_in[12];
  const float* cm_g  = (const float*)d_in[13];
  const float* cm_b  = (const float*)d_in[14];
  const float* w1    = (const float*)d_in[15];
  const float* b1    = (const float*)d_in[16];
  const float* w2    = (const float*)d_in[17];
  const float* b2    = (const float*)d_in[18];

  // workspace carve (~188 MB total)
  const size_t F32A = (size_t)NTOK * DD * sizeof(float);     // 33.5 MB
  const size_t F16A = (size_t)NTOK * DD * sizeof(_Float16);  // 16.8 MB
  const size_t WHALF = (size_t)LL * DD * DD;                 // halves / weight
  char* w = (char*)d_ws;
  float*    X    = (float*)w;                 w += F32A;
  float*    H    = (float*)w;                 w += F32A;
  float*    OP   = (float*)w;                 w += F32A;
  _Float16* H16  = (_Float16*)w;              w += F16A;
  _Float16* Q16  = (_Float16*)w;              w += F16A;
  _Float16* K16  = (_Float16*)w;              w += F16A;
  _Float16* V16  = (_Float16*)w;              w += F16A;  // swizzled-B layout
  _Float16* O16  = (_Float16*)w;              w += F16A;
  _Float16* wq16 = (_Float16*)w;              w += WHALF * 2;
  _Float16* wk16 = (_Float16*)w;              w += WHALF * 2;
  _Float16* wv16 = (_Float16*)w;              w += WHALF * 2;
  _Float16* wo16 = (_Float16*)w;              w += WHALF * 2;
  _Float16* w116 = (_Float16*)w;              w += WHALF * 2;
  _Float16* w216 = (_Float16*)w;              w += WHALF * 2;
  float*    CP   = (float*)w;                 /* S*S*4 = 1 MB */
  _Float16* C16  = K16;   // alias: K dead after attention
  _Float16* G16  = V16;   // alias: V dead after attention

  // one-time prep: swizzled weight conversion + relative-position bias table
  const int wn = (int)WHALF;
  const int wblk = (wn + 255) / 256;
  pack_bswz_kernel<<<wblk, 256, 0, stream>>>(wq, wq16, DD, DD, wn);
  pack_bswz_kernel<<<wblk, 256, 0, stream>>>(wk, wk16, DD, DD, wn);
  pack_bswz_kernel<<<wblk, 256, 0, stream>>>(wv, wv16, DD, DD, wn);
  pack_bswz_kernel<<<wblk, 256, 0, stream>>>(wo, wo16, DD, DD, wn);
  pack_bswz_kernel<<<wblk, 256, 0, stream>>>(w1, w116, DD, DD, wn);
  pack_bswz_kernel<<<wblk, 256, 0, stream>>>(w2, w216, DD, DD, wn);
  relpos_bias_kernel<<<(SS * SS + 255) / 256, 256, 0, stream>>>(table, CP, SS);

  // GEMM launch geometry: (NTOK/16)*(DD/64) wave tiles, 8 waves/block
  const int wtiles = (NTOK / 16) * (DD / 64);
  const int gblk = (wtiles * 32 + 255) / 256;

  for (int j = 0; j < LL; ++j) {
    const float* xin = (j == 0) ? x_in : X;
    const float* gj = ln_g + j * DD, *bj = ln_b + j * DD;
    const size_t wo_j = (size_t)j * DD * DD;

    // h = LN(x)
    ln_kernel<<<NTOK, DD, 0, stream>>>(xin, gj, bj, H, H16);

    // q/k = h @ W + b (f16 row-major); v in swizzled-B layout for P@V
    gemm_wmma_kernel<1, 0, 0><<<gblk, 256, 0, stream>>>(
        H16, wq16 + wo_j, bq + j * DD, nullptr, Q16, NTOK, DD, DD);
    gemm_wmma_kernel<1, 0, 0><<<gblk, 256, 0, stream>>>(
        H16, wk16 + wo_j, bk + j * DD, nullptr, K16, NTOK, DD, DD);
    gemm_wmma_kernel<2, 0, 0><<<gblk, 256, 0, stream>>>(
        H16, wv16 + wo_j, bv + j * DD, nullptr, V16, NTOK, DD, DD);

    // attention -> context (f16 row-major)
    attn_kernel<<<dim3(SS / 16, BB), 32, 0, stream>>>(
        Q16, K16, V16, CP, mask, O16, 1.0f / 16.0f /* 1/sqrt(D) */);

    // oproj = ctx @ wo + bo (f32)
    gemm_wmma_kernel<0, 0, 0><<<gblk, 256, 0, stream>>>(
        O16, wo16 + wo_j, bo + j * DD, nullptr, OP, NTOK, DD, DD);

    // x = h + LN_j(oproj); c = LN_cm(x)
    resid_ln_kernel<<<NTOK, DD, 0, stream>>>(
        H, OP, gj, bj, cm_g + j * DD, cm_b + j * DD, X, C16);

    // g = gelu(c @ w1 + b1) (f16)
    gemm_wmma_kernel<1, 1, 0><<<gblk, 256, 0, stream>>>(
        C16, w116 + wo_j, b1 + j * DD, nullptr, G16, NTOK, DD, DD);

    // x = x + g @ w2 + b2   (last layer -> d_out)
    float* dst = (j == LL - 1) ? (float*)d_out : X;
    gemm_wmma_kernel<0, 0, 1><<<gblk, 256, 0, stream>>>(
        G16, w216 + wo_j, b2 + j * DD, X, dst, NTOK, DD, DD);
  }
}
